// WindowAttention_86895778333421
// MI455X (gfx1250) — matrix operand order repaired and checked by reading, not compiled, not run on hardware
//
#include <hip/hip_runtime.h>

typedef __attribute__((ext_vector_type(16))) __bf16 v16bf;
typedef __attribute__((ext_vector_type(8)))  float  v8f;
typedef __attribute__((ext_vector_type(4)))  unsigned int u32x4;

#define B_   16
#define P_   256
#define N_   49
#define DIM_ 128
#define H_   4
#define HD_  32
#define SCALE_ 0.17677669529663687f  /* 32^-0.5 */

/* LDS strides in elements; multiples of 8 -> every fragment segment is a 16B
   aligned ds_load_b128 */
#define XS 136   /* x / q / k / o / w buffers: 128 + 8 pad */
#define VS 72    /* v-transposed buffer:        64 + 8 pad */
#define SS 72    /* score buffer (f32)                      */
#define AS 72    /* attn (bf16) buffer                      */

/* byte offsets into dynamic LDS (all 16B aligned) */
#define OFF_X   0
#define OFF_W   17408
#define OFF_Q   52224
#define OFF_K   69632
#define OFF_VT  87040
#define OFF_S   105472
#define OFF_A   123904
#define OFF_O   133120
#define LDS_BYTES 150528

#define WPRE_MAT   (DIM_ * XS)            /* halves per pre-padded weight matrix */
#define WPRE_BYTES ((size_t)4 * WPRE_MAT * 2)

struct alignas(8) us4 { unsigned short x, y, z, w; };

__device__ __forceinline__ unsigned short f2bf(float f) {
  __bf16 h = (__bf16)f;
  return __builtin_bit_cast(unsigned short, h);
}

union FragU { u32x4 q[2]; v16bf v; };

/* A fragment: 16x32 bf16, lane = M row; two contiguous 16B segments */
__device__ __forceinline__ v16bf load_a(const unsigned short* buf, int stride,
                                        int rowbase, int kbase, int lane) {
  const int g = lane >> 4, m = lane & 15;
  const unsigned short* p = buf + (rowbase + m) * stride + kbase + g * 8;
  FragU f;
  f.q[0] = *(const u32x4*)(p);        /* K = g*8 .. g*8+7        */
  f.q[1] = *(const u32x4*)(p + 16);   /* K = 16+g*8 .. 16+g*8+7  */
  return f.v;
}

/* B fragment: 32x16 bf16, lane = N column; buf holds columns of B as rows */
__device__ __forceinline__ v16bf load_b(const unsigned short* buf, int stride,
                                        int colbase, int kbase, int lane) {
  const int g = lane >> 4, n = lane & 15;
  const unsigned short* p = buf + (colbase + n) * stride + kbase + g * 8;
  FragU f;
  f.q[0] = *(const u32x4*)(p);        /* K = g*8 .. g*8+7        */
  f.q[1] = *(const u32x4*)(p + 16);   /* K = 16+g*8 .. 16+g*8+7  */
  return f.v;
}

__device__ __forceinline__ v8f wmma_bf16(v16bf a, v16bf b, v8f c) {
  return __builtin_amdgcn_wmma_f32_16x16x32_bf16(false, a, false, b,
                                                 (short)0, c, false, false);
}

/* one-time conversion of the four weight matrices to bf16, pre-padded to the
   XS row stride so the main kernel stages them with raw b128 copies */
__global__ __launch_bounds__(256)
void prep_weights(const float* __restrict__ Wq, const float* __restrict__ Wk,
                  const float* __restrict__ Wv, const float* __restrict__ Wp,
                  unsigned short* __restrict__ wout) {
  const int i = blockIdx.x * 256 + threadIdx.x;   /* 16384 threads total */
  const int m = i >> 12;                          /* matrix id (block-uniform) */
  const int r = i & 4095;
  const int o = r >> 5, d = (r & 31) * 4;
  const float* W = (m == 0) ? Wq : (m == 1) ? Wk : (m == 2) ? Wv : Wp;
  float4 val = *(const float4*)(W + o * DIM_ + d);
  us4 hv = { f2bf(val.x), f2bf(val.y), f2bf(val.z), f2bf(val.w) };
  *(us4*)(wout + m * WPRE_MAT + o * XS + d) = hv;
}

__global__ __launch_bounds__(256)
void winattn_fused(const float* __restrict__ q, const float* __restrict__ k,
                   const float* __restrict__ v, const float* __restrict__ bias,
                   const float* __restrict__ mask, const float* __restrict__ Wq,
                   const float* __restrict__ Wk, const float* __restrict__ Wv,
                   const float* __restrict__ Wp, const float* __restrict__ bp,
                   const unsigned short* __restrict__ wpre,
                   float* __restrict__ out) {
  extern __shared__ char smem[];
  unsigned short* xbuf  = (unsigned short*)(smem + OFF_X);
  unsigned short* wbuf  = (unsigned short*)(smem + OFF_W);
  unsigned short* qbuf  = (unsigned short*)(smem + OFF_Q);
  unsigned short* kbuf  = (unsigned short*)(smem + OFF_K);
  unsigned short* vbufT = (unsigned short*)(smem + OFF_VT);
  float*          sbuf  = (float*)         (smem + OFF_S);
  unsigned short* abuf  = (unsigned short*)(smem + OFF_A);
  unsigned short* obuf  = (unsigned short*)(smem + OFF_O);

  const int wg   = blockIdx.x;          /* b*P + p */
  const int p    = wg & (P_ - 1);
  const int tid  = threadIdx.x;
  const int wave = tid >> 5;
  const int lane = tid & 31;
  const size_t xoff = (size_t)wg * (N_ * DIM_);
  const size_t moff = (size_t)p  * (N_ * N_);

  /* stage activations: float4 global loads, ushort4 LDS stores, pad to 64 rows */
  auto stage_x = [&](const float* __restrict__ src) {
#pragma unroll 4
    for (int i = tid; i < 64 * (DIM_ / 4); i += 256) {
      int n = i >> 5, d = (i & 31) * 4;
      float4 val = make_float4(0.f, 0.f, 0.f, 0.f);
      if (n < N_) val = *(const float4*)(src + xoff + n * DIM_ + d);
      us4 hv = { f2bf(val.x), f2bf(val.y), f2bf(val.z), f2bf(val.w) };
      *(us4*)(xbuf + n * XS + d) = hv;
    }
  };
  /* stage weights: fast path = raw b128 copy of pre-converted bf16 */
  auto stage_w = [&](const float* __restrict__ W, int widx) {
    if (wpre) {
      const unsigned short* Wb = wpre + widx * WPRE_MAT;
#pragma unroll 4
      for (int i = tid; i < WPRE_MAT / 8; i += 256) {
        *(u32x4*)(wbuf + i * 8) = *(const u32x4*)(Wb + i * 8);
      }
    } else {
#pragma unroll 4
      for (int i = tid; i < DIM_ * (DIM_ / 4); i += 256) {
        int o = i >> 5, d = (i & 31) * 4;
        float4 val = *(const float4*)(W + o * DIM_ + d);
        us4 hv = { f2bf(val.x), f2bf(val.y), f2bf(val.z), f2bf(val.w) };
        *(us4*)(wbuf + o * XS + d) = hv;
      }
    }
  };

  /* C[64,128] = xbuf @ wbuf^T; per wave: rows tm..tm+15, cols tn0..tn0+63,
     A fragment shared across 4 column tiles */
  auto proj = [&](unsigned short* dst, bool transpose) {
    const int tm  = (wave >> 1) << 4;
    const int tn0 = (wave & 1) << 6;
    v8f c0 = {}, c1 = {}, c2 = {}, c3 = {};
#pragma unroll
    for (int kk = 0; kk < 4; ++kk) {
      v16bf a = load_a(xbuf, XS, tm, kk * 32, lane);
      c0 = wmma_bf16(a, load_b(wbuf, XS, tn0,      kk * 32, lane), c0);
      c1 = wmma_bf16(a, load_b(wbuf, XS, tn0 + 16, kk * 32, lane), c1);
      c2 = wmma_bf16(a, load_b(wbuf, XS, tn0 + 32, kk * 32, lane), c2);
      c3 = wmma_bf16(a, load_b(wbuf, XS, tn0 + 48, kk * 32, lane), c3);
    }
    const int colb = tn0 + (lane & 15);
    const int rb   = tm + ((lane >> 4) << 3);
#pragma unroll
    for (int r = 0; r < 8; ++r) {
      int row = rb + r;
      if (!transpose) {
        dst[row * XS + colb     ] = f2bf(c0[r]);
        dst[row * XS + colb + 16] = f2bf(c1[r]);
        dst[row * XS + colb + 32] = f2bf(c2[r]);
        dst[row * XS + colb + 48] = f2bf(c3[r]);
      } else {
        dst[(colb     ) * VS + row] = f2bf(c0[r]);
        dst[(colb + 16) * VS + row] = f2bf(c1[r]);
        dst[(colb + 32) * VS + row] = f2bf(c2[r]);
        dst[(colb + 48) * VS + row] = f2bf(c3[r]);
      }
    }
  };

  /* ---- QKV projections ---- */
  stage_w(Wq, 0); stage_x(q); __syncthreads();
  proj(qbuf, false);          __syncthreads();
  stage_w(Wk, 1); stage_x(k); __syncthreads();
  proj(kbuf, false);          __syncthreads();
  stage_w(Wv, 2); stage_x(v); __syncthreads();
  proj(vbufT, true);          __syncthreads();

  /* ---- per-head attention ---- */
  for (int h = 0; h < H_; ++h) {
    /* S = Qh @ Kh^T : [64,64], K=32; per wave one A, two B tiles */
    {
      const int tm  = (wave >> 1) << 4;
      const int tn0 = (wave & 1) << 5;
      v16bf a  = load_a(qbuf, XS, tm, h * HD_, lane);
      v8f c0 = {}, c1 = {};
      c0 = wmma_bf16(a, load_b(kbuf, XS, tn0,      h * HD_, lane), c0);
      c1 = wmma_bf16(a, load_b(kbuf, XS, tn0 + 16, h * HD_, lane), c1);
      const int colb = tn0 + (lane & 15);
      const int rb   = tm + ((lane >> 4) << 3);
#pragma unroll
      for (int r = 0; r < 8; ++r) {
        sbuf[(rb + r) * SS + colb     ] = c0[r];
        sbuf[(rb + r) * SS + colb + 16] = c1[r];
      }
    }
    __syncthreads();

    /* softmax: 2 threads per row, wave32 shuffle reductions */
    if (tid < 128) {
      const int i = tid >> 1, half = tid & 1;
      if (i < N_) {
        const int j0 = half ? 25 : 0;
        const int j1 = half ? N_ : 25;
        const float* brow = bias + ((size_t)h * N_ + i) * N_;
        const float* mrow = mask + moff + (size_t)i * N_;
        float mx = -3.0e38f;
        for (int j = j0; j < j1; ++j) {
          float val = (sbuf[i * SS + j] * SCALE_ + brow[j]) * mrow[j];
          sbuf[i * SS + j] = val;
          mx = fmaxf(mx, val);
        }
        mx = fmaxf(mx, __shfl_xor(mx, 1));
        float sum = 0.f;
        for (int j = j0; j < j1; ++j) {
          float e = __expf(sbuf[i * SS + j] - mx);
          sbuf[i * SS + j] = e;
          sum += e;
        }
        sum += __shfl_xor(sum, 1);
        float inv = 1.f / sum;
        for (int j = j0; j < j1; ++j) abuf[i * AS + j] = f2bf(sbuf[i * SS + j] * inv);
        if (half) {
          for (int j = N_; j < 64; ++j) abuf[i * AS + j] = 0;
        }
      } else if (i < 64) {
        for (int j = half * 32; j < half * 32 + 32; ++j) abuf[i * AS + j] = 0;
      }
    }
    __syncthreads();

    /* O_h = A @ Vh : [64,32], K=64 -> 2 WMMAs, one tile per wave */
    {
      const int tm = (wave >> 1) << 4, tn = (wave & 1) << 4;
      v8f c = {};
#pragma unroll
      for (int kk = 0; kk < 2; ++kk) {
        v16bf a = load_a(abuf, AS, tm, kk * 32, lane);
        v16bf b = load_b(vbufT, VS, h * HD_ + tn, kk * 32, lane);
        c = wmma_bf16(a, b, c);
      }
      const int colb = h * HD_ + tn + (lane & 15);
      const int rb   = tm + ((lane >> 4) << 3);
#pragma unroll
      for (int r = 0; r < 8; ++r) obuf[(rb + r) * XS + colb] = f2bf(c[r]);
    }
    __syncthreads();
  }

  /* ---- output projection: Y = O @ Wp^T + bp ---- */
  stage_w(Wp, 3);
  __syncthreads();
  {
    const int tm  = (wave >> 1) << 4;
    const int tn0 = (wave & 1) << 6;
    v8f c0 = {}, c1 = {}, c2 = {}, c3 = {};
#pragma unroll
    for (int kk = 0; kk < 4; ++kk) {
      v16bf a = load_a(obuf, XS, tm, kk * 32, lane);
      c0 = wmma_bf16(a, load_b(wbuf, XS, tn0,      kk * 32, lane), c0);
      c1 = wmma_bf16(a, load_b(wbuf, XS, tn0 + 16, kk * 32, lane), c1);
      c2 = wmma_bf16(a, load_b(wbuf, XS, tn0 + 32, kk * 32, lane), c2);
      c3 = wmma_bf16(a, load_b(wbuf, XS, tn0 + 48, kk * 32, lane), c3);
    }
    const int colb = tn0 + (lane & 15);
    const int rb   = tm + ((lane >> 4) << 3);
    const float b0 = bp[colb], b1 = bp[colb + 16];
    const float b2 = bp[colb + 32], b3 = bp[colb + 48];
#pragma unroll
    for (int r = 0; r < 8; ++r) {
      int row = rb + r;
      if (row < N_) {
        float* o = out + xoff + (size_t)row * DIM_ + colb;
        o[0]  = c0[r] + b0;
        o[16] = c1[r] + b1;
        o[32] = c2[r] + b2;
        o[48] = c3[r] + b3;
      }
    }
  }
}

extern "C" void kernel_launch(void* const* d_in, const int* in_sizes, int n_in,
                              void* d_out, int out_size, void* d_ws, size_t ws_size,
                              hipStream_t stream) {
  const float* q    = (const float*)d_in[0];
  const float* k    = (const float*)d_in[1];
  const float* v    = (const float*)d_in[2];
  const float* bias = (const float*)d_in[3];
  const float* mask = (const float*)d_in[4];
  const float* Wq   = (const float*)d_in[5];
  const float* Wk   = (const float*)d_in[6];
  const float* Wv   = (const float*)d_in[7];
  const float* Wp   = (const float*)d_in[8];
  const float* bp   = (const float*)d_in[9];
  float* out = (float*)d_out;

  const bool use_pre = (ws_size >= WPRE_BYTES);
  unsigned short* wpre = use_pre ? (unsigned short*)d_ws : nullptr;
  if (use_pre) {
    prep_weights<<<64, 256, 0, stream>>>(Wq, Wk, Wv, Wp, wpre);
  }

  dim3 grid(B_ * P_), block(256);
  winattn_fused<<<grid, block, LDS_BYTES, stream>>>(q, k, v, bias, mask,
                                                    Wq, Wk, Wv, Wp, bp, wpre, out);
  (void)in_sizes; (void)n_in; (void)out_size;
}
